// VAE_46883863003203
// MI455X (gfx1250) — compile-verified, matrix operand order checked by reading, not run on hardware
//
#include <hip/hip_runtime.h>

typedef __attribute__((ext_vector_type(16))) __bf16 v16bf;
typedef __attribute__((ext_vector_type(8)))  float  v8f;
typedef __attribute__((ext_vector_type(4)))  int    v4i;

#define BATCH   16384
#define LATENT  64
#define NEMB    8192

__device__ __forceinline__ unsigned short f2bf(float f) {
  unsigned int u = __float_as_uint(f);
  u += 0x7FFFu + ((u >> 16) & 1u);          // round-to-nearest-even
  return (unsigned short)(u >> 16);
}

union Frag { v16bf v; uint4 q[2]; };

// ---- gfx1250 async global->LDS copies (ASYNCcnt) -------------------------
// Builtin signatures (from clang-22 diagnostics):
//   b128: (v4i addrspace(1)*, v4i addrspace(3)*, int offset, int cpol)
//   b32 : (int addrspace(1)*, int addrspace(3)*, int offset, int cpol)
__device__ __forceinline__ void async_b128(const void* g, void* l) {
#if __has_builtin(__builtin_amdgcn_global_load_async_to_lds_b128)
  __builtin_amdgcn_global_load_async_to_lds_b128(
      (__attribute__((address_space(1))) v4i*)(uintptr_t)g,
      (__attribute__((address_space(3))) v4i*)(unsigned)(uintptr_t)l, 0, 0);
#else
  asm volatile("global_load_async_to_lds_b128 %0, %1, off"
               :: "v"((unsigned)(uintptr_t)l), "v"(g) : "memory");
#endif
}

__device__ __forceinline__ void async_b32(const void* g, void* l) {
#if __has_builtin(__builtin_amdgcn_global_load_async_to_lds_b32)
  __builtin_amdgcn_global_load_async_to_lds_b32(
      (__attribute__((address_space(1))) int*)(uintptr_t)g,
      (__attribute__((address_space(3))) int*)(unsigned)(uintptr_t)l, 0, 0);
#else
  asm volatile("global_load_async_to_lds_b32 %0, %1, off"
               :: "v"((unsigned)(uintptr_t)l), "v"(g) : "memory");
#endif
}

__device__ __forceinline__ void wait_async0() {
#if __has_builtin(__builtin_amdgcn_s_wait_asynccnt)
  __builtin_amdgcn_s_wait_asynccnt(0);
#else
  asm volatile("s_wait_asynccnt 0x0" ::: "memory");
#endif
}

// ---------------------------------------------------------------------------
// Fused GEMM: out[M,N] = act(A[M,K] @ W[N,K]^T + bias[N]); bf16 in, f32 acc.
// Double-buffered LDS tiles filled by async global->LDS DMA.
// ---------------------------------------------------------------------------
#define BM 128
#define BN 128
#define BK 32
#define LDT 40   // padded LDS stride (bf16 elems): conflict-free b128 reads

__global__ __launch_bounds__(256) void gemm_bias_act(
    const unsigned short* __restrict__ A,
    const unsigned short* __restrict__ W,
    const float* __restrict__ bias,
    unsigned short* __restrict__ out_bf,
    float* __restrict__ out_f32,
    int M, int N, int K, int act)
{
  __shared__ __align__(16) unsigned short As[2][BM * LDT];
  __shared__ __align__(16) unsigned short Ws[2][BN * LDT];

  const int tid   = threadIdx.x;
  const int baseM = blockIdx.y * BM;
  const int baseN = blockIdx.x * BN;
  const int wv    = tid >> 5;
  const int lane  = tid & 31;
  const int hf    = lane >> 4;
  const int l16   = lane & 15;
  const int waveM = (wv & 3) * 32;
  const int waveN = (wv >> 2) * 64;

  v8f acc[2][4];
  const v8f vz = {0.f,0.f,0.f,0.f,0.f,0.f,0.f,0.f};
#pragma unroll
  for (int m = 0; m < 2; ++m)
#pragma unroll
    for (int n = 0; n < 4; ++n) acc[m][n] = vz;

  auto stage = [&](int b, int k0) {
#pragma unroll
    for (int i = 0; i < 2; ++i) {
      int g  = tid + i * 256;           // 0..511 -> 128 rows x 4 groups of 8
      int r  = g >> 2;
      int c8 = (g & 3) * 8;
      int kk = k0 + c8;
      unsigned short* la = &As[b][r * LDT + c8];
      if (kk < K) async_b128(A + (size_t)(baseM + r) * K + kk, la);
      else       { uint4 z{}; *(uint4*)la = z; }
      unsigned short* lw = &Ws[b][r * LDT + c8];
      int wr = baseN + r;
      if (wr < N && kk < K) async_b128(W + (size_t)wr * K + kk, lw);
      else                  { uint4 z{}; *(uint4*)lw = z; }
    }
  };

  const int nt = (K + BK - 1) / BK;
  stage(0, 0);

  for (int kt = 0; kt < nt; ++kt) {
    const int cur = kt & 1;
    wait_async0();            // own async loads for tile kt complete
    __syncthreads();          // all waves loaded; prev readers of other buf done
    if (kt + 1 < nt) stage(cur ^ 1, (kt + 1) * BK);

    Frag a[2], b[4];
    const int kf = hf * 8;
    const int ks = 16 + hf * 8;
#pragma unroll
    for (int m = 0; m < 2; ++m) {
      int row = waveM + m * 16 + l16;
      a[m].q[0] = *(const uint4*)&As[cur][row * LDT + kf];
      a[m].q[1] = *(const uint4*)&As[cur][row * LDT + ks];
    }
#pragma unroll
    for (int n = 0; n < 4; ++n) {
      int row = waveN + n * 16 + l16;
      b[n].q[0] = *(const uint4*)&Ws[cur][row * LDT + kf];
      b[n].q[1] = *(const uint4*)&Ws[cur][row * LDT + ks];
    }
#pragma unroll
    for (int m = 0; m < 2; ++m)
#pragma unroll
      for (int n = 0; n < 4; ++n)
        acc[m][n] = __builtin_amdgcn_wmma_f32_16x16x32_bf16(
            false, a[m].v, false, b[n].v, (short)0, acc[m][n], false, false);
  }

  const bool wbf = (out_bf  != nullptr);
  const bool wf  = (out_f32 != nullptr);
#pragma unroll
  for (int n = 0; n < 4; ++n) {
    int col = baseN + waveN + n * 16 + l16;
    if (col < N) {
      float bv = bias[col];
#pragma unroll
      for (int m = 0; m < 2; ++m) {
        int rbase = baseM + waveM + m * 16 + hf * 8;
#pragma unroll
        for (int e = 0; e < 8; ++e) {
          float v = acc[m][n][e] + bv;
          if (act == 1)      v = fmaxf(v, 0.f);
          else if (act == 2) v = 1.f / (1.f + __expf(-v));
          size_t idx = (size_t)(rbase + e) * N + col;
          if (wbf) out_bf[idx]  = f2bf(v);
          if (wf)  out_f32[idx] = v;
        }
      }
    }
  }
}

// ---------------------------------------------------------------------------
// Fused VQ: 128 z-rows resident, stream codebook tiles (double-buffered async),
// WMMA scores, running argmin of (|e|^2 - 2 z.e), gather zq, per-block SSE.
// ---------------------------------------------------------------------------
#define VLD 72

__global__ __launch_bounds__(256) void vq_kernel(
    const unsigned short* __restrict__ zbf,
    const float* __restrict__ zf,
    const unsigned short* __restrict__ ewbf,
    const float* __restrict__ ewf,
    const float* __restrict__ e2,
    unsigned short* __restrict__ zq,
    float* __restrict__ blocksum)
{
  __shared__ __align__(16) unsigned short Zs[128 * VLD];
  __shared__ __align__(16) unsigned short Es[2][128 * VLD];
  __shared__ float e2s[2][128];
  __shared__ float minv_s[8 * 32 * 8];
  __shared__ int   mini_s[8 * 32 * 8];
  __shared__ float sred[256];

  const int tid   = threadIdx.x;
  const int baseM = blockIdx.x * 128;
  const int wv    = tid >> 5;
  const int lane  = tid & 31;
  const int hf    = lane >> 4;
  const int l16   = lane & 15;

  auto stageE = [&](int b, int blk) {
#pragma unroll
    for (int i = 0; i < 4; ++i) {
      int g = tid + i * 256;
      int r = g >> 3, c8 = (g & 7) * 8;
      async_b128(ewbf + (size_t)(blk + r) * LATENT + c8, &Es[b][r * VLD + c8]);
    }
    if (tid < 128) async_b32(e2 + blk + tid, &e2s[b][tid]);
  };

#pragma unroll
  for (int i = 0; i < 4; ++i) {           // stage z tile (128 x 64 bf16)
    int g = tid + i * 256;
    int r = g >> 3, c8 = (g & 7) * 8;
    async_b128(zbf + (size_t)(baseM + r) * LATENT + c8, &Zs[r * VLD + c8]);
  }
  stageE(0, 0);
  wait_async0();
  __syncthreads();

  const int waveM = wv * 16;              // each wave owns 16 z-rows
  Frag a0, a1;
  {
    int row = waveM + l16;
    a0.q[0] = *(const uint4*)&Zs[row * VLD + (hf * 8)];
    a0.q[1] = *(const uint4*)&Zs[row * VLD + (16 + hf * 8)];
    a1.q[0] = *(const uint4*)&Zs[row * VLD + (32 + hf * 8)];
    a1.q[1] = *(const uint4*)&Zs[row * VLD + (48 + hf * 8)];
  }

  float minv[8];
  int   mini[8];
#pragma unroll
  for (int e = 0; e < 8; ++e) { minv[e] = 3.4e38f; mini[e] = 0; }

  const int NT = NEMB / 128;
  for (int t = 0; t < NT; ++t) {
    const int cur = t & 1;
    if (t) { wait_async0(); __syncthreads(); }
    if (t + 1 < NT) stageE(cur ^ 1, (t + 1) * 128);

#pragma unroll
    for (int n = 0; n < 8; ++n) {
      Frag b0, b1;
      int row = n * 16 + l16;
      b0.q[0] = *(const uint4*)&Es[cur][row * VLD + (hf * 8)];
      b0.q[1] = *(const uint4*)&Es[cur][row * VLD + (16 + hf * 8)];
      b1.q[0] = *(const uint4*)&Es[cur][row * VLD + (32 + hf * 8)];
      b1.q[1] = *(const uint4*)&Es[cur][row * VLD + (48 + hf * 8)];
      v8f acc = {0.f,0.f,0.f,0.f,0.f,0.f,0.f,0.f};
      acc = __builtin_amdgcn_wmma_f32_16x16x32_bf16(
          false, a0.v, false, b0.v, (short)0, acc, false, false);
      acc = __builtin_amdgcn_wmma_f32_16x16x32_bf16(
          false, a1.v, false, b1.v, (short)0, acc, false, false);
      int   colL = n * 16 + l16;
      float ec   = e2s[cur][colL];
      int   gidx = t * 128 + colL;
#pragma unroll
      for (int e = 0; e < 8; ++e) {
        float d = ec - 2.0f * acc[e];     // |z|^2 is row-constant: drop it
        if (d < minv[e]) { minv[e] = d; mini[e] = gidx; }
      }
    }
  }

  {
    int base = (wv * 32 + lane) * 8;
#pragma unroll
    for (int e = 0; e < 8; ++e) { minv_s[base + e] = minv[e]; mini_s[base + e] = mini[e]; }
  }
  __syncthreads();

  float sse = 0.f;
  if (tid < 128) {
    int row = tid;
    int w = row >> 4, r = row & 15;
    int lb = (r < 8) ? 0 : 16;
    int e  = r & 7;
    float best = 3.4e38f; int bidx = 1 << 30;
    for (int l = 0; l < 16; ++l) {
      int o = (w * 32 + lb + l) * 8 + e;
      float v  = minv_s[o];
      int   ii = mini_s[o];
      if (v < best || (v == best && ii < bidx)) { best = v; bidx = ii; }
    }
    const float* ev = ewf + (size_t)bidx * LATENT;           // full-precision gather
    const float* zr = zf  + (size_t)(baseM + row) * LATENT;
    unsigned short* zo = zq + (size_t)(baseM + row) * LATENT;
    for (int d = 0; d < LATENT; ++d) {
      float q  = ev[d];
      float df = q - zr[d];
      sse += df * df;
      zo[d] = f2bf(q);
    }
  }
  sred[tid] = sse;
  __syncthreads();
  for (int s = 128; s > 0; s >>= 1) {
    if (tid < s) sred[tid] += sred[tid + s];
    __syncthreads();
  }
  if (tid == 0) blocksum[blockIdx.x] = sred[0];   // deterministic partials
}

// ---------------------------------------------------------------------------
__global__ void cvt_kernel(const float* __restrict__ src,
                           unsigned short* __restrict__ dst, int n) {
  int i = blockIdx.x * 256 + threadIdx.x;
  if (i < n) dst[i] = f2bf(src[i]);
}

__global__ void e2_kernel(const float* __restrict__ ewf, float* __restrict__ e2) {
  int i = blockIdx.x * 256 + threadIdx.x;
  if (i < NEMB) {
    const float* p = ewf + (size_t)i * LATENT;
    float s = 0.f;
    for (int d = 0; d < LATENT; ++d) s += p[d] * p[d];
    e2[i] = s;
  }
}

__global__ void fin_kernel(const float* __restrict__ blocksum, float* __restrict__ out) {
  if (threadIdx.x == 0 && blockIdx.x == 0) {
    float s = 0.f;
    for (int i = 0; i < BATCH / 128; ++i) s += blocksum[i];
    out[0] = 1.25f * s / (float)(BATCH * LATENT);   // 0.25*commit + embed
  }
}

// ---------------------------------------------------------------------------
extern "C" void kernel_launch(void* const* d_in, const int* in_sizes, int n_in,
                              void* d_out, int out_size, void* d_ws, size_t ws_size,
                              hipStream_t stream) {
  (void)in_sizes; (void)n_in; (void)out_size; (void)ws_size;
  const float* x   = (const float*)d_in[0];
  const float* ew  = (const float*)d_in[1];
  const float* ew1 = (const float*)d_in[2];   const float* eb1 = (const float*)d_in[3];
  const float* ew2 = (const float*)d_in[4];   const float* eb2 = (const float*)d_in[5];
  const float* ew3 = (const float*)d_in[6];   const float* eb3 = (const float*)d_in[7];
  const float* ew4 = (const float*)d_in[8];   const float* eb4 = (const float*)d_in[9];
  const float* dw1 = (const float*)d_in[10];  const float* db1 = (const float*)d_in[11];
  const float* dw2 = (const float*)d_in[12];  const float* db2 = (const float*)d_in[13];
  const float* dw3 = (const float*)d_in[14];  const float* db3 = (const float*)d_in[15];
  const float* dw4 = (const float*)d_in[16];  const float* db4 = (const float*)d_in[17];

  char* ws = (char*)d_ws;
  size_t off = 0;
  auto alloc = [&](size_t bytes) -> size_t {
    size_t o = off; off += (bytes + 255) & ~(size_t)255; return o;
  };
  const size_t o_xb  = alloc((size_t)BATCH * 784 * 2);
  const size_t o_w1  = alloc((size_t)512 * 784 * 2);
  const size_t o_w2  = alloc((size_t)256 * 512 * 2);
  const size_t o_w3  = alloc((size_t)128 * 256 * 2);
  const size_t o_w4  = alloc((size_t)64  * 128 * 2);
  const size_t o_dw1 = alloc((size_t)128 * 64  * 2);
  const size_t o_dw2 = alloc((size_t)256 * 128 * 2);
  const size_t o_dw3 = alloc((size_t)512 * 256 * 2);
  const size_t o_dw4 = alloc((size_t)784 * 512 * 2);
  const size_t o_ewb = alloc((size_t)NEMB * LATENT * 2);
  const size_t o_e2  = alloc((size_t)NEMB * 4);
  const size_t o_ls  = alloc((size_t)(BATCH / 128) * 4);
  const size_t o_h1  = alloc((size_t)BATCH * 512 * 2);    // reused as d3
  const size_t o_h2  = alloc((size_t)BATCH * 256 * 2);    // reused as d2
  const size_t o_h3  = alloc((size_t)BATCH * 128 * 2);    // reused as d1
  const size_t o_zbf = alloc((size_t)BATCH * LATENT * 2);
  const size_t o_zf  = alloc((size_t)BATCH * LATENT * 4);
  const size_t o_zq  = alloc((size_t)BATCH * LATENT * 2);

  auto U16 = [&](size_t o) { return (unsigned short*)(ws + o); };
  auto F32 = [&](size_t o) { return (float*)(ws + o); };

  auto cvt = [&](const float* src, size_t dst, size_t n) {
    cvt_kernel<<<dim3((unsigned)((n + 255) / 256)), dim3(256), 0, stream>>>(
        src, U16(dst), (int)n);
  };
  cvt(x,   o_xb,  (size_t)BATCH * 784);
  cvt(ew1, o_w1,  512 * 784);
  cvt(ew2, o_w2,  256 * 512);
  cvt(ew3, o_w3,  128 * 256);
  cvt(ew4, o_w4,  64  * 128);
  cvt(dw1, o_dw1, 128 * 64);
  cvt(dw2, o_dw2, 256 * 128);
  cvt(dw3, o_dw3, 512 * 256);
  cvt(dw4, o_dw4, 784 * 512);
  cvt(ew,  o_ewb, NEMB * LATENT);
  e2_kernel<<<dim3((NEMB + 255) / 256), dim3(256), 0, stream>>>(ew, F32(o_e2));

  auto gemm = [&](size_t a, size_t w, const float* b,
                  unsigned short* obf, float* of32, int N, int K, int act) {
    dim3 grid((N + BN - 1) / BN, BATCH / BM);
    gemm_bias_act<<<grid, dim3(256), 0, stream>>>(
        U16(a), U16(w), b, obf, of32, BATCH, N, K, act);
  };

  // encoder
  gemm(o_xb, o_w1, eb1, U16(o_h1), nullptr,    512, 784, 1);
  gemm(o_h1, o_w2, eb2, U16(o_h2), nullptr,    256, 512, 1);
  gemm(o_h2, o_w3, eb3, U16(o_h3), nullptr,    128, 256, 1);
  gemm(o_h3, o_w4, eb4, U16(o_zbf), F32(o_zf),  64, 128, 0);
  // vector quantization
  vq_kernel<<<dim3(BATCH / 128), dim3(256), 0, stream>>>(
      U16(o_zbf), F32(o_zf), U16(o_ewb), ew, F32(o_e2), U16(o_zq), F32(o_ls));
  // decoder (recon straight to d_out, sigmoid fused)
  float* recon = (float*)d_out;
  gemm(o_zq, o_dw1, db1, U16(o_h3), nullptr,   128, 64,  1);
  gemm(o_h3, o_dw2, db2, U16(o_h2), nullptr,   256, 128, 1);
  gemm(o_h2, o_dw3, db3, U16(o_h1), nullptr,   512, 256, 1);
  gemm(o_h1, o_dw4, db4, nullptr,   recon,     784, 512, 2);
  // scalar vq_loss at d_out[BATCH*784]
  fin_kernel<<<dim3(1), dim3(32), 0, stream>>>(F32(o_ls), recon + (size_t)BATCH * 784);
}
